// HeadAttentionMulti_48172353191943
// MI455X (gfx1250) — matrix operand order repaired and checked by reading, not compiled, not run on hardware
//
#include <hip/hip_runtime.h>

#define BATCH 2
#define SEQ   2048
#define NH    8
#define DH    64
#define DM    512   // NH*DH

typedef __bf16 bf16;
typedef __attribute__((ext_vector_type(16))) __bf16 v16bf;
typedef __attribute__((ext_vector_type(8)))  __bf16 bf16x8;
typedef __attribute__((ext_vector_type(8)))  float  v8f;
typedef __attribute__((ext_vector_type(4)))  float  f32x4;

__device__ __forceinline__ v8f wmma_bf16(v16bf a, v16bf b, v8f c) {
  // D = A(16x32 bf16) * B(32x16 bf16) + C(16x16 f32)
  return __builtin_amdgcn_wmma_f32_16x16x32_bf16(false, a, false, b, (short)0, c,
                                                 false, false);
}

__device__ __forceinline__ v16bf cat8(bf16x8 lo, bf16x8 hi) {
  return __builtin_shufflevector(lo, hi, 0, 1, 2, 3, 4, 5, 6, 7,
                                 8, 9, 10, 11, 12, 13, 14, 15);
}

__device__ __forceinline__ v16bf cvt16(f32x4 x0, f32x4 x1, f32x4 x2, f32x4 x3) {
  v16bf r;
#pragma unroll
  for (int j = 0; j < 4; ++j) {
    r[j]      = (bf16)x0[j];
    r[4 + j]  = (bf16)x1[j];
    r[8 + j]  = (bf16)x2[j];
    r[12 + j] = (bf16)x3[j];
  }
  return r;
}

// ---- A fragment (16x32 M x K). Lane L holds row M=L%16;
// K elements {hi..hi+7} and {16+hi..16+hi+7}, hi = (L>=16)*8.  All 16B aligned.
__device__ __forceinline__ v16bf load_a_f32(const float* __restrict__ base, int ld,
                                            int row0, int col0, int lane) {
  const float* p = base + (size_t)(row0 + (lane & 15)) * ld + col0 + ((lane >> 4) * 8);
  return cvt16(*(const f32x4*)p, *(const f32x4*)(p + 4),
               *(const f32x4*)(p + 16), *(const f32x4*)(p + 20));
}

__device__ __forceinline__ v16bf load_a_f32_sum(const float* __restrict__ b0,
                                                const float* __restrict__ b1, int ld,
                                                int row0, int col0, int lane) {
  const size_t off = (size_t)(row0 + (lane & 15)) * ld + col0 + ((lane >> 4) * 8);
  const float* p = b0 + off;
  const float* q = b1 + off;
  f32x4 x0 = *(const f32x4*)p        + *(const f32x4*)q;
  f32x4 x1 = *(const f32x4*)(p + 4)  + *(const f32x4*)(q + 4);
  f32x4 x2 = *(const f32x4*)(p + 16) + *(const f32x4*)(q + 16);
  f32x4 x3 = *(const f32x4*)(p + 20) + *(const f32x4*)(q + 20);
  return cvt16(x0, x1, x2, x3);
}

__device__ __forceinline__ v16bf load_a_bf16(const bf16* __restrict__ base, int ld,
                                             int row0, int col0, int lane) {
  const bf16* p = base + (size_t)(row0 + (lane & 15)) * ld + col0 + ((lane >> 4) * 8);
  return cat8(*(const bf16x8*)p, *(const bf16x8*)(p + 16));
}

// ---- B fragment (32x16 K x N): lane L holds column N=L%16;
// K elements {hi..hi+7} and {16+hi..16+hi+7}, hi = (L>=16)*8.
__device__ __forceinline__ v16bf load_b_f32(const float* __restrict__ base, int ld,
                                            int r0, int c0, int lane) {
  const float* p = base + (size_t)(r0 + ((lane >> 4) * 8)) * ld + c0 + (lane & 15);
  v16bf r;
#pragma unroll
  for (int j = 0; j < 8; ++j) {
    r[j]     = (bf16)p[(size_t)j * ld];
    r[8 + j] = (bf16)p[(size_t)(16 + j) * ld];
  }
  return r;
}

__device__ __forceinline__ v16bf load_b_bf16(const bf16* __restrict__ base, int ld,
                                             int r0, int c0, int lane) {
  const bf16* p = base + (size_t)(r0 + ((lane >> 4) * 8)) * ld + c0 + (lane & 15);
  v16bf r;
#pragma unroll
  for (int j = 0; j < 8; ++j) {
    r[j]     = p[(size_t)j * ld];
    r[8 + j] = p[(size_t)(16 + j) * ld];
  }
  return r;
}

__device__ __forceinline__ int wave_gid() {
  return __builtin_amdgcn_readfirstlane(blockIdx.x * (blockDim.x >> 5) +
                                        (threadIdx.x >> 5));
}

// =====================================================================
// Kernel 1: fused projections; one wave computes a 16x64 output strip.
// job 0: Qs = (X+P)Wq + 2bq  -> row-major bf16
// job 1: Qc = X Wq + bq      -> row-major bf16
// job 2: Kc = X Wk + bk      -> transposed bf16  KcT[b][h][d][s]
// job 3: Kr = P Wk + bk      -> transposed bf16  KrT[b][h][d][s]
// job 4: Vc = X Wv + bv      -> row-major bf16
// =====================================================================
__global__ void proj_kernel(const float* __restrict__ X, const float* __restrict__ P,
                            const float* __restrict__ Wq, const float* __restrict__ bq,
                            const float* __restrict__ Wk, const float* __restrict__ bk,
                            const float* __restrict__ Wv, const float* __restrict__ bv,
                            bf16* __restrict__ Qs, bf16* __restrict__ Qc,
                            bf16* __restrict__ KcT, bf16* __restrict__ KrT,
                            bf16* __restrict__ Vc) {
  const int lane = threadIdx.x & 31;
  const int gid  = wave_gid();
  const int GROUPS = (BATCH * SEQ / 16) * (DM / 64);  // 2048 strips per job
  const int job = gid / GROUPS;
  const int rem = gid % GROUPS;
  const int m0  = (rem / (DM / 64)) * 16;
  const int n0  = (rem % (DM / 64)) * 64;

  const float* W    = (job <= 1) ? Wq : (job <= 3 ? Wk : Wv);
  const float* bias = (job <= 1) ? bq : (job <= 3 ? bk : bv);

  v8f acc[4];
#pragma unroll
  for (int t = 0; t < 4; ++t) acc[t] = (v8f){0.f, 0.f, 0.f, 0.f, 0.f, 0.f, 0.f, 0.f};

  if (job == 0) {
#pragma unroll 2
    for (int k0 = 0; k0 < DM; k0 += 32) {
      v16bf a = load_a_f32_sum(X, P, DM, m0, k0, lane);
#pragma unroll
      for (int t = 0; t < 4; ++t)
        acc[t] = wmma_bf16(a, load_b_f32(W, DM, k0, n0 + 16 * t, lane), acc[t]);
    }
  } else {
    const float* A = (job == 3) ? P : X;
#pragma unroll 2
    for (int k0 = 0; k0 < DM; k0 += 32) {
      v16bf a = load_a_f32(A, DM, m0, k0, lane);
#pragma unroll
      for (int t = 0; t < 4; ++t)
        acc[t] = wmma_bf16(a, load_b_f32(W, DM, k0, n0 + 16 * t, lane), acc[t]);
    }
  }

  const int hi = (lane >> 4) * 8;
  const float bscale = (job == 0) ? 2.0f : 1.0f;

  if (job == 2 || job == 3) {
    bf16* dst = (job == 2) ? KcT : KrT;
    const int batch = m0 / SEQ;
    const int s0    = (m0 % SEQ) + hi;
#pragma unroll
    for (int t = 0; t < 4; ++t) {
      const int col = n0 + 16 * t + (lane & 15);
      const float bb = bias[col] * bscale;
      const int h = col >> 6, d = col & (DH - 1);
      bf16x8 v;
#pragma unroll
      for (int r = 0; r < 8; ++r) v[r] = (bf16)(acc[t][r] + bb);
      *(bf16x8*)(dst + (((size_t)(batch * NH + h) * DH + d) * SEQ + s0)) = v;
    }
  } else {
    bf16* dst = (job == 0) ? Qs : (job == 1 ? Qc : Vc);
#pragma unroll
    for (int t = 0; t < 4; ++t) {
      const int col = n0 + 16 * t + (lane & 15);
      const float bb = bias[col] * bscale;
#pragma unroll
      for (int r = 0; r < 8; ++r)
        dst[(size_t)(m0 + hi + r) * DM + col] = (bf16)(acc[t][r] + bb);
    }
  }
}

// =====================================================================
// Kernel 2: logits = (Qs.KcT + Qc.KrT)/8 -> attn (f32).
// One wave: 1 q-tile x 8 k-tiles; A fragments live in registers (32 WMMAs).
// =====================================================================
__global__ void logits_kernel(const bf16* __restrict__ Qs, const bf16* __restrict__ Qc,
                              const bf16* __restrict__ KcT, const bf16* __restrict__ KrT,
                              float* __restrict__ attn) {
  const int lane = threadIdx.x & 31;
  const int gid  = wave_gid();
  const int QT = SEQ / 16;        // 128
  const int KG = SEQ / (16 * 8);  // 16 k-groups of 8 tiles
  const int bh  = gid / (QT * KG);
  const int rem = gid % (QT * KG);
  const int q0  = (rem / KG) * 16;
  const int kg0 = (rem % KG) * 128;
  const int b = bh / NH, h = bh % NH;

  const bf16* qs  = Qs + (size_t)b * SEQ * DM;
  const bf16* qc  = Qc + (size_t)b * SEQ * DM;
  const bf16* kct = KcT + (size_t)bh * DH * SEQ;
  const bf16* krt = KrT + (size_t)bh * DH * SEQ;

  const v16bf aqs0 = load_a_bf16(qs, DM, q0, h * DH + 0, lane);
  const v16bf aqs1 = load_a_bf16(qs, DM, q0, h * DH + 32, lane);
  const v16bf aqc0 = load_a_bf16(qc, DM, q0, h * DH + 0, lane);
  const v16bf aqc1 = load_a_bf16(qc, DM, q0, h * DH + 32, lane);

  const int hi = (lane >> 4) * 8;
  float* dst = attn + ((size_t)bh * SEQ + q0 + hi) * SEQ + (lane & 15);

#pragma unroll
  for (int kt = 0; kt < 8; ++kt) {
    const int k0 = kg0 + kt * 16;
    v8f acc = {0.f, 0.f, 0.f, 0.f, 0.f, 0.f, 0.f, 0.f};
    acc = wmma_bf16(aqs0, load_b_bf16(kct, SEQ, 0, k0, lane), acc);
    acc = wmma_bf16(aqs1, load_b_bf16(kct + 32 * SEQ, SEQ, 0, k0, lane), acc);
    acc = wmma_bf16(aqc0, load_b_bf16(krt, SEQ, 0, k0, lane), acc);
    acc = wmma_bf16(aqc1, load_b_bf16(krt + 32 * SEQ, SEQ, 0, k0, lane), acc);
#pragma unroll
    for (int r = 0; r < 8; ++r) dst[(size_t)r * SEQ + k0] = acc[r] * 0.125f;
  }
}

// =====================================================================
// Kernel 3: row softmax in place (one block of 256 per row).
// =====================================================================
__global__ void softmax_kernel(float* __restrict__ attn) {
  __shared__ float red[256];
  const int tid = threadIdx.x;
  float* p = attn + (size_t)blockIdx.x * SEQ;

  float v[SEQ / 256];
  float m = -3.0e38f;
#pragma unroll
  for (int j = 0; j < SEQ / 256; ++j) {
    v[j] = p[tid + j * 256];
    m = fmaxf(m, v[j]);
  }
  red[tid] = m;
  __syncthreads();
  for (int s = 128; s > 0; s >>= 1) {
    if (tid < s) red[tid] = fmaxf(red[tid], red[tid + s]);
    __syncthreads();
  }
  m = red[0];
  __syncthreads();

  float sum = 0.f;
#pragma unroll
  for (int j = 0; j < SEQ / 256; ++j) {
    v[j] = __expf(v[j] - m);
    sum += v[j];
  }
  red[tid] = sum;
  __syncthreads();
  for (int s = 128; s > 0; s >>= 1) {
    if (tid < s) red[tid] += red[tid + s];
    __syncthreads();
  }
  const float inv = 1.0f / red[0];
#pragma unroll
  for (int j = 0; j < SEQ / 256; ++j) p[tid + j * 256] = v[j] * inv;
}

// =====================================================================
// Kernel 4: Z = attn @ Vc. One wave: 1 q-tile x all 4 d-tiles (DH=64).
// Each attn A fragment (f32->bf16) feeds 4 WMMAs.
// =====================================================================
__global__ void av_kernel(const float* __restrict__ attn, const bf16* __restrict__ Vc,
                          float* __restrict__ Z) {
  const int lane = threadIdx.x & 31;
  const int gid  = wave_gid();
  const int QT = SEQ / 16;  // 128
  const int bh = gid / QT;
  const int q0 = (gid % QT) * 16;
  const int b = bh / NH, h = bh % NH;

  const float* arow = attn + (size_t)bh * SEQ * SEQ;  // [SEQ][SEQ]
  const bf16*  vrow = Vc + (size_t)b * SEQ * DM;      // [SEQ][DM]

  v8f acc[4];
#pragma unroll
  for (int t = 0; t < 4; ++t) acc[t] = (v8f){0.f, 0.f, 0.f, 0.f, 0.f, 0.f, 0.f, 0.f};

  for (int k0 = 0; k0 < SEQ; k0 += 32) {
    v16bf a = load_a_f32(arow, SEQ, q0, k0, lane);
#pragma unroll
    for (int t = 0; t < 4; ++t)
      acc[t] = wmma_bf16(a, load_b_bf16(vrow, DM, k0, h * DH + 16 * t, lane), acc[t]);
  }

  const int hi = (lane >> 4) * 8;
#pragma unroll
  for (int t = 0; t < 4; ++t) {
    const int col = h * DH + 16 * t + (lane & 15);
#pragma unroll
    for (int r = 0; r < 8; ++r)
      Z[(size_t)(b * SEQ + q0 + hi + r) * DM + col] = acc[t][r];
  }
}

// =====================================================================
// Kernel 5: out = Z @ Wo + bo. One wave computes a 16x64 strip.
// =====================================================================
__global__ void out_kernel(const float* __restrict__ Z, const float* __restrict__ Wo,
                           const float* __restrict__ bo, float* __restrict__ out) {
  const int lane = threadIdx.x & 31;
  const int gid  = wave_gid();
  const int m0 = (gid / (DM / 64)) * 16;
  const int n0 = (gid % (DM / 64)) * 64;

  v8f acc[4];
#pragma unroll
  for (int t = 0; t < 4; ++t) acc[t] = (v8f){0.f, 0.f, 0.f, 0.f, 0.f, 0.f, 0.f, 0.f};

#pragma unroll 2
  for (int k0 = 0; k0 < DM; k0 += 32) {
    v16bf a = load_a_f32(Z, DM, m0, k0, lane);
#pragma unroll
    for (int t = 0; t < 4; ++t)
      acc[t] = wmma_bf16(a, load_b_f32(Wo, DM, k0, n0 + 16 * t, lane), acc[t]);
  }

  const int hi = (lane >> 4) * 8;
#pragma unroll
  for (int t = 0; t < 4; ++t) {
    const int col = n0 + 16 * t + (lane & 15);
    const float bb = bo[col];
#pragma unroll
    for (int r = 0; r < 8; ++r)
      out[(size_t)(m0 + hi + r) * DM + col] = acc[t][r] + bb;
  }
}

// =====================================================================
extern "C" void kernel_launch(void* const* d_in, const int* in_sizes, int n_in,
                              void* d_out, int out_size, void* d_ws, size_t ws_size,
                              hipStream_t stream) {
  const float* X  = (const float*)d_in[0];  // emb_x
  const float* P  = (const float*)d_in[1];  // pe_rel_t
  const float* Wq = (const float*)d_in[2];
  const float* bq = (const float*)d_in[3];
  const float* Wk = (const float*)d_in[4];
  const float* bk = (const float*)d_in[5];
  const float* Wv = (const float*)d_in[6];
  const float* bv = (const float*)d_in[7];
  const float* Wo = (const float*)d_in[8];
  const float* bo = (const float*)d_in[9];

  float* out  = (float*)d_out;                   // [B, S, DM]
  float* attn = out + (size_t)BATCH * SEQ * DM;  // [B, H, S, S]

  const size_t NPROJ = (size_t)BATCH * SEQ * DM;  // 2,097,152 elems
  bf16* Qs  = (bf16*)d_ws;
  bf16* Qc  = Qs + NPROJ;
  bf16* KcT = Qc + NPROJ;
  bf16* KrT = KcT + NPROJ;
  bf16* Vc  = KrT + NPROJ;
  float* Z  = (float*)(Vc + NPROJ);  // total ws use: 28 MB

  // 1) projections: 5 jobs * 2048 strips = 10240 waves (8 waves/block)
  proj_kernel<<<1280, 256, 0, stream>>>(X, P, Wq, bq, Wk, bk, Wv, bv,
                                        Qs, Qc, KcT, KrT, Vc);
  // 2) logits: 16 bh * 128 q-tiles * 16 k-groups = 32768 waves
  logits_kernel<<<4096, 256, 0, stream>>>(Qs, Qc, KcT, KrT, attn);
  // 3) softmax: one block per (b,h,q) row
  softmax_kernel<<<BATCH * NH * SEQ, 256, 0, stream>>>(attn);
  // 4) attn @ V: 16 bh * 128 q-tiles = 2048 waves
  av_kernel<<<256, 256, 0, stream>>>(attn, Vc, Z);
  // 5) output projection: 256 m-tiles * 8 n-groups = 2048 waves
  out_kernel<<<256, 256, 0, stream>>>(Z, Wo, bo, out);
}
